// sLSTMBlock_54485955117298
// MI455X (gfx1250) — matrix operand order repaired and checked by reading, not compiled, not run on hardware
//
#include <hip/hip_runtime.h>
#include <cstdint>

#define Bb 4
#define Tt 4096
#define Dd 1024
#define Kk 4
#define Pp 1364
#define PAD 1376            // P padded to multiple of 32 (K/N padding for GEMMs)
#define PR 1408             // gate-weight rows padded (covers 88 N-tiles of 16)
#define BT (Bb * Tt)
#define CAPv 15.0f

typedef __attribute__((ext_vector_type(16))) __bf16 v16bf;
typedef __attribute__((ext_vector_type(8)))  float  v8f;

// ---------------------------------------------------------------------------
// Async global->LDS staging (CDNA5 GLOBAL_LOAD_ASYNC_TO_LDS_B128), guarded so
// the file compiles on toolchains without the builtins.
// ---------------------------------------------------------------------------
#if defined(__has_builtin)
#if __has_builtin(__builtin_amdgcn_global_load_async_to_lds_b128) && \
    __has_builtin(__builtin_amdgcn_s_wait_asynccnt)
#define ASYNC_LDS 1
#endif
#endif

#ifdef ASYNC_LDS
// builtin params are int-vector(4) pointers in AS1 (global) / AS3 (LDS)
typedef int vs4i __attribute__((vector_size(16)));
typedef __attribute__((address_space(1))) vs4i* glb_v4p;
typedef __attribute__((address_space(3))) vs4i* lds_v4p;
#endif

__device__ __forceinline__ void cp16(void* l, const void* g) {
#ifdef ASYNC_LDS
    __builtin_amdgcn_global_load_async_to_lds_b128(
        (glb_v4p)(uintptr_t)g, (lds_v4p)(unsigned)(uintptr_t)l, 0, 0);
#else
    *(uint4*)l = *(const uint4*)g;
#endif
}
__device__ __forceinline__ void cp_wait() {
#ifdef ASYNC_LDS
    __builtin_amdgcn_s_wait_asynccnt(0);
#endif
}

// ---------------------------------------------------------------------------
// WMMA fragment helpers (contiguous k = 16*(lane/16) + s in A and B, wave32)
// ---------------------------------------------------------------------------
__device__ __forceinline__ v16bf frag_A_rowmajor(const __bf16* tile /*16x32 stride 32*/, int lane) {
    int m  = lane & 15;
    int kb = (lane >> 4) << 4;
    const __bf16* p0 = tile + m * 32 + kb;
    const __bf16* p1 = p0 + 8;
    v16bf a;
#pragma unroll
    for (int i = 0; i < 8; ++i) { a[i] = p0[i]; a[i + 8] = p1[i]; }
    return a;
}
__device__ __forceinline__ v16bf frag_B_nk(const __bf16* tile /*[n][k] 16x32*/, int lane) {
    int n  = lane & 15;
    int kb = (lane >> 4) << 4;
    const __bf16* p = tile + n * 32 + kb;
    v16bf b;
#pragma unroll
    for (int i = 0; i < 16; ++i) b[i] = p[i];
    return b;
}
__device__ __forceinline__ v8f wmma_bf16(v16bf a, v16bf b, v8f c) {
    return __builtin_amdgcn_wmma_f32_16x16x32_bf16(false, a, false, b, (short)0, c, false, false);
}
__device__ __forceinline__ float sigmoid_f(float v) { return 1.0f / (1.0f + __expf(-v)); }
__device__ __forceinline__ float softcap_f(float z) { return CAPv * tanhf(z * (1.0f / CAPv)); }

// ---------------------------------------------------------------------------
// 1) RMSNorm -> bf16
// ---------------------------------------------------------------------------
__global__ __launch_bounds__(256) void k_rmsnorm(const float* __restrict__ x,
                                                 const float* __restrict__ w,
                                                 __bf16* __restrict__ xn) {
    __shared__ float red[8];
    int row = blockIdx.x;
    const float* xr = x + (size_t)row * Dd;
    float s = 0.f;
    for (int i = threadIdx.x; i < Dd; i += 256) { float v = xr[i]; s += v * v; }
#pragma unroll
    for (int o = 16; o > 0; o >>= 1) s += __shfl_down(s, o);
    if ((threadIdx.x & 31) == 0) red[threadIdx.x >> 5] = s;
    __syncthreads();
    float tot = red[0] + red[1] + red[2] + red[3] + red[4] + red[5] + red[6] + red[7];
    float rn = rsqrtf(tot * (1.0f / (float)Dd) + 1e-5f);
    for (int i = threadIdx.x; i < Dd; i += 256)
        xn[(size_t)row * Dd + i] = (__bf16)(w[i] * xr[i] * rn);
}

// ---------------------------------------------------------------------------
// 2) Weight preprocessing
// ---------------------------------------------------------------------------
// conv_w [K][Din][Dout] fp32 -> convT [K][Dout][Din] bf16
__global__ void k_convT(const float* __restrict__ s, __bf16* __restrict__ d) {
    for (int i = blockIdx.x * blockDim.x + threadIdx.x; i < Kk * Dd * Dd;
         i += gridDim.x * blockDim.x) {
        int o = i % Dd; int t = i / Dd; int din = t % Dd; int k = t / Dd;
        d[((size_t)k * Dd + o) * Dd + din] = (__bf16)s[i];
    }
}
// cast + zero-pad trailing rows: src [rv][cols] -> dst [rp][cols]
__global__ void k_castpad_rows(const float* __restrict__ s, __bf16* __restrict__ d,
                               int rv, int rp, int cols) {
    long n = (long)rp * cols;
    for (long i = blockIdx.x * blockDim.x + threadIdx.x; i < n;
         i += (long)gridDim.x * blockDim.x) {
        long r = i / cols, c = i % cols;
        d[i] = (r < rv) ? (__bf16)s[r * cols + c] : (__bf16)0.f;
    }
}
// cast + zero-pad trailing cols: src [rows][cv] -> dst [rows][cp]
__global__ void k_castpad_cols(const float* __restrict__ s, __bf16* __restrict__ d,
                               int rows, int cv, int cp) {
    long n = (long)rows * cp;
    for (long i = blockIdx.x * blockDim.x + threadIdx.x; i < n;
         i += (long)gridDim.x * blockDim.x) {
        long r = i / cp, c = i % cp;
        d[i] = (c < cv) ? (__bf16)s[r * cv + c] : (__bf16)0.f;
    }
}

// ---------------------------------------------------------------------------
// 3) Causal conv as GEMM.  Block: M=32, N=256 (8 waves x 32 cols).
//    Wave tile: 2 M-frags x 2 N-frags x 4 taps = 16 WMMA per 32-wide K step.
// ---------------------------------------------------------------------------
__global__ __launch_bounds__(256) void k_conv_gemm(const __bf16* __restrict__ xn,
                                                   const __bf16* __restrict__ convT,
                                                   __bf16* __restrict__ xc) {
    __shared__ __align__(16) __bf16 As[35 * 32];          // slab rows t0-3 .. t0+31
    __shared__ __align__(16) __bf16 Bs[8][Kk][2][512];
    int wave = threadIdx.x >> 5, lane = threadIdx.x & 31;
    int m0 = blockIdx.x * 32;
    int bidx = m0 / Tt, t0 = m0 % Tt;
    int o0 = (blockIdx.y * 8 + wave) * 32;
    v8f acc[2][2] = {};

    for (int d0 = 0; d0 < Dd; d0 += 32) {
        __syncthreads();
        // A slab: masked (t<0 zero-fill), regular ds staging
        for (int i = threadIdx.x; i < 35 * 4; i += 256) {
            int row = i >> 2, c16 = i & 3;
            int tt = t0 + row - (Kk - 1);
            uint4 z = {0u, 0u, 0u, 0u};
            if (tt >= 0)
                z = *(const uint4*)(xn + ((size_t)bidx * Tt + tt) * Dd + d0 + c16 * 8);
            *(uint4*)(As + row * 32 + c16 * 8) = z;
        }
        // B tiles: always in-bounds -> async staging (16 chunks per lane)
#pragma unroll
        for (int j = 0; j < 16; ++j) {
            int c = lane + j * 32;
            int k = c >> 7, r = c & 127;
            int nt = r >> 6, n = (r >> 2) & 15, c16 = r & 3;
            cp16(&Bs[wave][k][nt][n * 32 + c16 * 8],
                 convT + ((size_t)k * Dd + (o0 + nt * 16 + n)) * Dd + d0 + c16 * 8);
        }
        cp_wait();
        __syncthreads();
#pragma unroll
        for (int k = 0; k < Kk; ++k) {
            v16bf b0 = frag_B_nk(&Bs[wave][k][0][0], lane);
            v16bf b1 = frag_B_nk(&Bs[wave][k][1][0], lane);
            v16bf a0 = frag_A_rowmajor(As + k * 32, lane);          // M-tile 0, tap k
            v16bf a1 = frag_A_rowmajor(As + (16 + k) * 32, lane);   // M-tile 1, tap k
            acc[0][0] = wmma_bf16(a0, b0, acc[0][0]);
            acc[0][1] = wmma_bf16(a0, b1, acc[0][1]);
            acc[1][0] = wmma_bf16(a1, b0, acc[1][0]);
            acc[1][1] = wmma_bf16(a1, b1, acc[1][1]);
        }
    }
    int n = lane & 15, mlo = (lane >> 4) * 8;
#pragma unroll
    for (int mt = 0; mt < 2; ++mt)
#pragma unroll
        for (int nt = 0; nt < 2; ++nt)
#pragma unroll
            for (int r = 0; r < 8; ++r)
                xc[((size_t)(m0 + mt * 16 + mlo + r)) * Dd + o0 + nt * 16 + n] =
                    (__bf16)acc[mt][nt][r];
}

// ---------------------------------------------------------------------------
// 4) Gate GEMM.  Block: M=64, wave -> one 16-wide P tile, 4 gates.
//    Wave tile: 4 M-frags x 4 gates = 16 WMMA per K step (8 frag loads).
// ---------------------------------------------------------------------------
__global__ __launch_bounds__(256) void k_gate_gemm(const __bf16* __restrict__ xc,
                                                   const __bf16* __restrict__ Wg, // [4][PR][D]
                                                   __bf16* __restrict__ f_tm,
                                                   __bf16* __restrict__ g_tm,
                                                   __bf16* __restrict__ o_tm) {
    __shared__ __align__(16) __bf16 As[64 * 32];
    __shared__ __align__(16) __bf16 Bs[8][4][512];
    int wave = threadIdx.x >> 5, lane = threadIdx.x & 31;
    int m0 = blockIdx.x * 64;
    int bidx = m0 / Tt, t0 = m0 % Tt;
    int p0 = (blockIdx.y * 8 + wave) * 16;       // < PR always (88 tiles max)
    v8f acc[4][4] = {};

    for (int d0 = 0; d0 < Dd; d0 += 32) {
        __syncthreads();
        {   // shared A tile 64x32: exactly one 16B chunk per thread
            int row = threadIdx.x >> 2, c16 = threadIdx.x & 3;
            cp16(As + row * 32 + c16 * 8,
                 xc + ((size_t)(m0 + row)) * Dd + d0 + c16 * 8);
        }
        // per-wave B tiles for 4 gates (rows padded/zeroed -> unguarded)
#pragma unroll
        for (int j = 0; j < 8; ++j) {
            int c = lane + j * 32;
            int g = c >> 6, n = (c >> 2) & 15, c16 = c & 3;
            cp16(&Bs[wave][g][n * 32 + c16 * 8],
                 Wg + ((size_t)g * PR + (p0 + n)) * Dd + d0 + c16 * 8);
        }
        cp_wait();
        __syncthreads();
        v16bf a0 = frag_A_rowmajor(As + 0 * 512, lane);
        v16bf a1 = frag_A_rowmajor(As + 1 * 512, lane);
        v16bf a2 = frag_A_rowmajor(As + 2 * 512, lane);
        v16bf a3 = frag_A_rowmajor(As + 3 * 512, lane);
#pragma unroll
        for (int g = 0; g < 4; ++g) {
            v16bf b = frag_B_nk(&Bs[wave][g][0], lane);
            acc[g][0] = wmma_bf16(a0, b, acc[g][0]);
            acc[g][1] = wmma_bf16(a1, b, acc[g][1]);
            acc[g][2] = wmma_bf16(a2, b, acc[g][2]);
            acc[g][3] = wmma_bf16(a3, b, acc[g][3]);
        }
    }
    int n = lane & 15, p = p0 + n, mlo = (lane >> 4) * 8;
    if (p < Pp) {
#pragma unroll
        for (int mt = 0; mt < 4; ++mt)
#pragma unroll
            for (int r = 0; r < 8; ++r) {
                float zi = acc[0][mt][r], zf = acc[1][mt][r];
                float zo = acc[2][mt][r], zc = acc[3][mt][r];
                float ig = sigmoid_f(softcap_f(zi));
                float fg = sigmoid_f(softcap_f(zf));
                float og = sigmoid_f(softcap_f(zo));
                float gg = ig * tanhf(zc);
                int t = t0 + mt * 16 + mlo + r;
                size_t idx = ((size_t)t * Bb + bidx) * PAD + p;   // time-major
                f_tm[idx] = (__bf16)fg;
                g_tm[idx] = (__bf16)gg;
                o_tm[idx] = (__bf16)og;
            }
    }
}

// ---------------------------------------------------------------------------
// 5) Gated recurrence (one thread per (b,p); pad threads zero y)
// ---------------------------------------------------------------------------
__global__ __launch_bounds__(256) void k_scan(const __bf16* __restrict__ f_tm,
                                              const __bf16* __restrict__ g_tm,
                                              const __bf16* __restrict__ o_tm,
                                              const float* __restrict__ h0,
                                              __bf16* __restrict__ y_tm,
                                              float* __restrict__ h_out) {
    int tid = blockIdx.x * 256 + threadIdx.x;
    if (tid >= Bb * PAD) return;
    int b = tid / PAD, p = tid % PAD;
    const size_t stride = (size_t)Bb * PAD;
    size_t off = tid;
    if (p >= Pp) {   // zero padded y columns so the out-GEMM K-padding is exact
        for (int t = 0; t < Tt; ++t, off += stride) y_tm[off] = (__bf16)0.f;
        return;
    }
    float h = h0[b * Pp + p];
    for (int t = 0; t < Tt; ++t, off += stride) {
        __builtin_prefetch(f_tm + off + 16 * stride, 0, 1);
        __builtin_prefetch(g_tm + off + 16 * stride, 0, 1);
        __builtin_prefetch(o_tm + off + 16 * stride, 0, 1);
        float f = (float)f_tm[off], g = (float)g_tm[off], o = (float)o_tm[off];
        h = __builtin_fmaf(f, h, g);
        y_tm[off] = (__bf16)(o * tanhf(h));
    }
    h_out[b * Pp + p] = h;
}

// ---------------------------------------------------------------------------
// 6) Output GEMM + residual.  Block: M=64, N=256.  Wave: 4 M x 2 N frags.
//    K runs over padded P (43 full 32-steps, zero-padded tails).
// ---------------------------------------------------------------------------
__global__ __launch_bounds__(256) void k_out_gemm(const __bf16* __restrict__ y_tm,
                                                  const __bf16* __restrict__ WoutB, // [D][PAD]
                                                  const float* __restrict__ x,
                                                  float* __restrict__ out) {
    __shared__ __align__(16) __bf16 As[64 * 32];
    __shared__ __align__(16) __bf16 Bs[8][2][512];
    int wave = threadIdx.x >> 5, lane = threadIdx.x & 31;
    int m0 = blockIdx.x * 64;
    int bidx = m0 / Tt, t0 = m0 % Tt;
    int d0 = (blockIdx.y * 8 + wave) * 32;
    v8f acc[2][4] = {};

    for (int p0 = 0; p0 < PAD; p0 += 32) {
        __syncthreads();
        {   // A tile: rows (t0+row, bidx) of y_tm, contiguous in padded p
            int row = threadIdx.x >> 2, c16 = threadIdx.x & 3;
            cp16(As + row * 32 + c16 * 8,
                 y_tm + ((size_t)(t0 + row) * Bb + bidx) * PAD + p0 + c16 * 8);
        }
#pragma unroll
        for (int j = 0; j < 4; ++j) {
            int c = lane + j * 32;
            int nt = c >> 6, n = (c >> 2) & 15, c16 = c & 3;
            cp16(&Bs[wave][nt][n * 32 + c16 * 8],
                 WoutB + (size_t)(d0 + nt * 16 + n) * PAD + p0 + c16 * 8);
        }
        cp_wait();
        __syncthreads();
        v16bf a0 = frag_A_rowmajor(As + 0 * 512, lane);
        v16bf a1 = frag_A_rowmajor(As + 1 * 512, lane);
        v16bf a2 = frag_A_rowmajor(As + 2 * 512, lane);
        v16bf a3 = frag_A_rowmajor(As + 3 * 512, lane);
#pragma unroll
        for (int nt = 0; nt < 2; ++nt) {
            v16bf b = frag_B_nk(&Bs[wave][nt][0], lane);
            acc[nt][0] = wmma_bf16(a0, b, acc[nt][0]);
            acc[nt][1] = wmma_bf16(a1, b, acc[nt][1]);
            acc[nt][2] = wmma_bf16(a2, b, acc[nt][2]);
            acc[nt][3] = wmma_bf16(a3, b, acc[nt][3]);
        }
    }
    int n = lane & 15, mlo = (lane >> 4) * 8;
#pragma unroll
    for (int nt = 0; nt < 2; ++nt)
#pragma unroll
        for (int mt = 0; mt < 4; ++mt)
#pragma unroll
            for (int r = 0; r < 8; ++r) {
                size_t o = ((size_t)(m0 + mt * 16 + mlo + r)) * Dd + d0 + nt * 16 + n;
                out[o] = x[o] + acc[nt][mt][r];
            }
}

// ---------------------------------------------------------------------------
extern "C" void kernel_launch(void* const* d_in, const int* in_sizes, int n_in,
                              void* d_out, int out_size, void* d_ws, size_t ws_size,
                              hipStream_t stream) {
    (void)in_sizes; (void)n_in; (void)out_size; (void)ws_size;
    const float* x     = (const float*)d_in[0];
    const float* h0    = (const float*)d_in[1];
    const float* rmsw  = (const float*)d_in[2];
    const float* convw = (const float*)d_in[3];
    const float* Wi    = (const float*)d_in[4];
    const float* Wf    = (const float*)d_in[5];
    const float* Wo    = (const float*)d_in[6];
    const float* Wc    = (const float*)d_in[7];
    const float* Wout  = (const float*)d_in[8];
    float* out   = (float*)d_out;
    float* hlast = out + (size_t)BT * Dd;

    char* ws = (char*)d_ws;
    size_t off = 0;
    auto take = [&](size_t bytes) -> char* {
        char* p = ws + off;
        off = (off + bytes + 255) & ~(size_t)255;
        return p;
    };
    const size_t tmP = (size_t)Tt * Bb * PAD;
    __bf16* xn    = (__bf16*)take((size_t)BT * Dd * 2);
    __bf16* xc    = (__bf16*)take((size_t)BT * Dd * 2);
    __bf16* convT = (__bf16*)take((size_t)Kk * Dd * Dd * 2);
    __bf16* Wg    = (__bf16*)take((size_t)4 * PR * Dd * 2);
    __bf16* WoutB = (__bf16*)take((size_t)Dd * PAD * 2);
    __bf16* f_tm  = (__bf16*)take(tmP * 2);
    __bf16* g_tm  = (__bf16*)take(tmP * 2);
    __bf16* o_tm  = (__bf16*)take(tmP * 2);
    __bf16* y_tm  = (__bf16*)take(tmP * 2);

    k_rmsnorm<<<BT, 256, 0, stream>>>(x, rmsw, xn);
    k_convT<<<2048, 256, 0, stream>>>(convw, convT);
    k_castpad_rows<<<2048, 256, 0, stream>>>(Wi, Wg + (size_t)0 * PR * Dd, Pp, PR, Dd);
    k_castpad_rows<<<2048, 256, 0, stream>>>(Wf, Wg + (size_t)1 * PR * Dd, Pp, PR, Dd);
    k_castpad_rows<<<2048, 256, 0, stream>>>(Wo, Wg + (size_t)2 * PR * Dd, Pp, PR, Dd);
    k_castpad_rows<<<2048, 256, 0, stream>>>(Wc, Wg + (size_t)3 * PR * Dd, Pp, PR, Dd);
    k_castpad_cols<<<2048, 256, 0, stream>>>(Wout, WoutB, Dd, Pp, PAD);

    k_conv_gemm<<<dim3(BT / 32, Dd / 256), 256, 0, stream>>>(xn, convT, xc);
    k_gate_gemm<<<dim3(BT / 64, 11), 256, 0, stream>>>(xc, Wg, f_tm, g_tm, o_tm);
    k_scan<<<(Bb * PAD + 255) / 256, 256, 0, stream>>>(f_tm, g_tm, o_tm, h0, y_tm, hlast);
    k_out_gemm<<<dim3(BT / 64, Dd / 256), 256, 0, stream>>>(y_tm, WoutB, x, out);
}
